// InsideDecoder_4294967296690
// MI455X (gfx1250) — matrix operand order repaired and checked by reading, not compiled, not run on hardware
//
#include <hip/hip_runtime.h>
#include <hip/hip_bf16.h>
#include <math.h>

// ---------------------------------------------------------------------------
// InsideDecoder on MI455X (gfx1250, wave32, WMMA).
//
// Pipeline:
//   rel  = tanh(pf[N,32] @ Wn[32,48])            (VALU, 0.15 GFLOP)
//   rows = [rel3 | feat128] for 800k rows
//   h1   = relu(rows @ W1[131,128])   \
//   h2   = relu(h1   @ W2[128,128])    >  66 GFLOP -> v_wmma_f32_16x16x32_f16
//   fc3  = relu(h2   @ W3[128, 64])   /
// Roofline: ~250 MB traffic -> ~11 us @ 23.3 TB/s; f16 WMMA keeps the GEMM
// chain near that floor (fp32 VALU would be ~40x slower than the HBM time).
// ---------------------------------------------------------------------------

typedef __attribute__((ext_vector_type(16))) _Float16 v16h;
typedef __attribute__((ext_vector_type(8)))  float    v8f;

union Frag16 { v16h h; uint4 q[2]; };   // 32 bytes = 16 halves

#define NPTS   50000
#define NB     16
#define KPAD   160      // 131 padded to 5 chunks of 32
#define WAVES  4        // waves (points) per block

// ---------------------------------------------------------------------------
// Kernel 0: weights -> transposed (out-major) f16, W1 padded K: 131 -> 160.
//   w1t[n][k] = W1[k][n] (k<131, else 0)   n<128, k<160
//   w2t[n][k] = W2[k][n]                   n<128, k<128
//   w3t[n][k] = W3[k][n]                   n<64,  k<128
// ---------------------------------------------------------------------------
__global__ void wcvt_kernel(const float* __restrict__ W1,
                            const float* __restrict__ W2,
                            const float* __restrict__ W3,
                            _Float16* __restrict__ w1t,
                            _Float16* __restrict__ w2t,
                            _Float16* __restrict__ w3t) {
  int tid = blockIdx.x * blockDim.x + threadIdx.x;
  if (tid < 128 * KPAD) {
    int n = tid / KPAD, k = tid % KPAD;
    w1t[tid] = (k < 131) ? (_Float16)W1[k * 128 + n] : (_Float16)0.0f;
  } else if (tid < 128 * KPAD + 128 * 128) {
    int t = tid - 128 * KPAD;
    int n = t >> 7, k = t & 127;
    w2t[t] = (_Float16)W2[k * 128 + n];
  } else if (tid < 128 * KPAD + 128 * 128 + 64 * 128) {
    int t = tid - (128 * KPAD + 128 * 128);
    int n = t >> 7, k = t & 127;
    w3t[t] = (_Float16)W3[k * 64 + n];
  }
}

// ---------------------------------------------------------------------------
// Kernel 1: rel = tanh(pf @ Wn + bn); also output_points = rel * 0.25.
// One thread per (point, j<48). out_points flat index (p*16+m)*3+k == p*48+j.
// ---------------------------------------------------------------------------
__global__ void rel_kernel(const float* __restrict__ features,
                           const float* __restrict__ Wn,
                           const float* __restrict__ bn,
                           _Float16* __restrict__ rel16,
                           float* __restrict__ out_points) {
  int tid = blockIdx.x * blockDim.x + threadIdx.x;
  if (tid >= NPTS * 48) return;
  int p = tid / 48, j = tid % 48;
  const float* pf = features + (long)p * 160;   // first 32 cols = point feats
  float acc = bn[j];
#pragma unroll
  for (int i = 0; i < 32; ++i) acc = fmaf(pf[i], Wn[i * 48 + j], acc);
  float t = tanhf(acc);
  rel16[tid]      = (_Float16)t;
  out_points[tid] = t * 0.25f;
}

// ---------------------------------------------------------------------------
// Kernel 1b: feat -> f16 (once per point, not 16x), and output_batch ids.
// Batch ids are int32 bit patterns in the concatenated output buffer.
// ---------------------------------------------------------------------------
__global__ void feat_kernel(const float* __restrict__ features,
                            _Float16* __restrict__ feat16,
                            int* __restrict__ batch_out) {
  int tid = blockIdx.x * blockDim.x + threadIdx.x;
  if (tid >= NPTS * 128) return;
  int p = tid >> 7, c = tid & 127;
  feat16[tid] = (_Float16)features[(long)p * 160 + 32 + c];
  if (c < NB) batch_out[p * NB + c] = p;
}

// ---------------------------------------------------------------------------
// Kernel 2: the WMMA MLP. One wave = one point = one 16-row M-tile.
// Wave32 fragment layouts (CDNA5 ISA 7.12.2):
//   A (16x32 f16): lane -> M = lane&15; halves = X[M][Kb..Kb+7] ++ X[M][Kb+16..Kb+23],
//                  Kb = kc*32 + 8*(lane>=16)
//   B (32x16 f16): lane -> N = lane&15 (of transposed weights);
//                  halves = Wt[N][Kb..Kb+7] ++ Wt[N][Kb+16..Kb+23], Kb as for A
//   C/D (16x16 f32): lane -> N = lane&15; VGPR v -> M = v + 8*(lane>=16)
// ---------------------------------------------------------------------------
__global__ __launch_bounds__(WAVES * 32) void
mlp_kernel(const _Float16* __restrict__ rel16,
           const _Float16* __restrict__ feat16,
           const _Float16* __restrict__ w1t,
           const _Float16* __restrict__ w2t,
           const _Float16* __restrict__ w3t,
           const float* __restrict__ b1,
           const float* __restrict__ b2,
           const float* __restrict__ b3,
           float* __restrict__ fc3_out) {
  __shared__ __align__(16) _Float16 Xt[WAVES][16 * KPAD];  // 20 KB
  __shared__ __align__(16) _Float16 H1[WAVES][16 * 128];   // 16 KB
  __shared__ __align__(16) _Float16 H2[WAVES][16 * 128];   // 16 KB

  const int wave = threadIdx.x >> 5;
  const int lane = threadIdx.x & 31;
  const int p    = blockIdx.x * WAVES + wave;   // 12500 * 4 == 50000 exactly
  const int m16  = lane & 15;
  const int hi   = lane >> 4;

  _Float16* xt = Xt[wave];
  _Float16* h1 = H1[wave];
  _Float16* h2 = H2[wave];

  // ---- Build input tile X: 16 rows x 160 cols = [rel3 | feat128 | 0...] ----
  {
    const _Float16* relp = rel16  + (long)p * 48;
    const _Float16* fp   = feat16 + (long)p * 128;
    for (int idx = lane; idx < 16 * KPAD; idx += 32) {  // 80 uniform iters
      int row = idx / KPAD, c = idx % KPAD;
      _Float16 v;
      if (c < 3)        v = relp[row * 3 + c];
      else if (c < 131) v = fp[c - 3];
      else              v = (_Float16)0.0f;
      xt[idx] = v;
    }
  }

  // ---- Layer 1: 16x160 @ 160x128, bias+ReLU fused -> H1 ----
  for (int nb = 0; nb < 8; ++nb) {
    const int n = nb * 16 + m16;
    float bias = b1[n];
    v8f acc;
#pragma unroll
    for (int i = 0; i < 8; ++i) acc[i] = bias;
#pragma unroll
    for (int kc = 0; kc < 5; ++kc) {
      Frag16 a, b;
      const int ka = kc * 32 + hi * 8;
      a.q[0] = *(const uint4*)(xt + m16 * KPAD + ka);
      a.q[1] = *(const uint4*)(xt + m16 * KPAD + ka + 16);
      const _Float16* bp = w1t + (long)n * KPAD + kc * 32 + hi * 8;
      b.q[0] = *(const uint4*)(bp);
      b.q[1] = *(const uint4*)(bp + 16);
      acc = __builtin_amdgcn_wmma_f32_16x16x32_f16(
          false, a.h, false, b.h, (short)0, acc, false, false);
    }
#pragma unroll
    for (int v = 0; v < 8; ++v)
      h1[(v + hi * 8) * 128 + n] = (_Float16)fmaxf(acc[v], 0.0f);
  }

  // ---- Layer 2: 16x128 @ 128x128 -> H2 ----
  for (int nb = 0; nb < 8; ++nb) {
    const int n = nb * 16 + m16;
    float bias = b2[n];
    v8f acc;
#pragma unroll
    for (int i = 0; i < 8; ++i) acc[i] = bias;
#pragma unroll
    for (int kc = 0; kc < 4; ++kc) {
      Frag16 a, b;
      const int ka = kc * 32 + hi * 8;
      a.q[0] = *(const uint4*)(h1 + m16 * 128 + ka);
      a.q[1] = *(const uint4*)(h1 + m16 * 128 + ka + 16);
      const _Float16* bp = w2t + (long)n * 128 + kc * 32 + hi * 8;
      b.q[0] = *(const uint4*)(bp);
      b.q[1] = *(const uint4*)(bp + 16);
      acc = __builtin_amdgcn_wmma_f32_16x16x32_f16(
          false, a.h, false, b.h, (short)0, acc, false, false);
    }
#pragma unroll
    for (int v = 0; v < 8; ++v)
      h2[(v + hi * 8) * 128 + n] = (_Float16)fmaxf(acc[v], 0.0f);
  }

  // ---- Layer 3: 16x128 @ 128x64 -> global fc3 (f32), ReLU fused ----
  for (int nb = 0; nb < 4; ++nb) {
    const int n = nb * 16 + m16;
    float bias = b3[n];
    v8f acc;
#pragma unroll
    for (int i = 0; i < 8; ++i) acc[i] = bias;
#pragma unroll
    for (int kc = 0; kc < 4; ++kc) {
      Frag16 a, b;
      const int ka = kc * 32 + hi * 8;
      a.q[0] = *(const uint4*)(h2 + m16 * 128 + ka);
      a.q[1] = *(const uint4*)(h2 + m16 * 128 + ka + 16);
      const _Float16* bp = w3t + (long)n * 128 + kc * 32 + hi * 8;
      b.q[0] = *(const uint4*)(bp);
      b.q[1] = *(const uint4*)(bp + 16);
      acc = __builtin_amdgcn_wmma_f32_16x16x32_f16(
          false, a.h, false, b.h, (short)0, acc, false, false);
    }
#pragma unroll
    for (int v = 0; v < 8; ++v) {
      const long row = (long)p * 16 + v + hi * 8;
      fc3_out[row * 64 + n] = fmaxf(acc[v], 0.0f);
    }
  }
}

// ---------------------------------------------------------------------------
// Launcher. Workspace layout (bytes, 16B-aligned chunks; ~17.7 MB total):
//   [0)          rel16  50000*48  f16   4,800,000
//   [4800000)    feat16 50000*128 f16  12,800,000
//   [17600000)   w1t    128*160   f16      40,960
//   [17640960)   w2t    128*128   f16      32,768
//   [17673728)   w3t     64*128   f16      16,384
// Output (flat, ref return order):
//   [0)          output_points 800000*3 f32
//   [2400000)    fc3           800000*64 f32
//   [53600000)   output_batch  800000 int32 (bit pattern)
// ---------------------------------------------------------------------------
extern "C" void kernel_launch(void* const* d_in, const int* in_sizes, int n_in,
                              void* d_out, int out_size, void* d_ws, size_t ws_size,
                              hipStream_t stream) {
  const float* features = (const float*)d_in[0];
  const float* Wn = (const float*)d_in[1];
  const float* bn = (const float*)d_in[2];
  const float* W1 = (const float*)d_in[3];
  const float* b1 = (const float*)d_in[4];
  const float* W2 = (const float*)d_in[5];
  const float* b2 = (const float*)d_in[6];
  const float* W3 = (const float*)d_in[7];
  const float* b3 = (const float*)d_in[8];

  char* ws = (char*)d_ws;
  _Float16* rel16  = (_Float16*)(ws);
  _Float16* feat16 = (_Float16*)(ws + 4800000);
  _Float16* w1t    = (_Float16*)(ws + 17600000);
  _Float16* w2t    = (_Float16*)(ws + 17640960);
  _Float16* w3t    = (_Float16*)(ws + 17673728);

  float* out        = (float*)d_out;
  float* out_points = out;                 //  2,400,000 f32
  float* fc3        = out + 2400000;       // 51,200,000 f32
  int*   batch      = (int*)(out + 53600000);  // 800,000 i32

  wcvt_kernel<<<(128 * KPAD + 128 * 128 + 64 * 128 + 255) / 256, 256, 0, stream>>>(
      W1, W2, W3, w1t, w2t, w3t);
  rel_kernel<<<(NPTS * 48) / 256, 256, 0, stream>>>(
      features, Wn, bn, rel16, out_points);
  feat_kernel<<<(NPTS * 128) / 256, 256, 0, stream>>>(
      features, feat16, batch);
  mlp_kernel<<<NPTS / WAVES, WAVES * 32, 0, stream>>>(
      rel16, feat16, w1t, w2t, w3t, b1, b2, b3, fc3);
}
